// SNNLanguageModel_71210557767976
// MI455X (gfx1250) — matrix operand order repaired and checked by reading, not compiled, not run on hardware
//
#include <hip/hip_runtime.h>
#include <cmath>

typedef __attribute__((ext_vector_type(16))) _Float16 v16h;
typedef __attribute__((ext_vector_type(8)))  float    v8f;

namespace cfg {
constexpr int Bb = 8, Ss = 64, Dd = 256, DN = 1024, NBk = 2, Kk = 8;
constexpr int VOC = 50257, VOCP = 50272, NDTILE = VOCP / 16;   // 3142 decoder tiles
// ---- f16 weight region (element offsets into d_ws interpreted as _Float16*) ----
constexpr size_t SZ_XW = (size_t)DN * Dd;    // 1024x256 transposed x-weights
constexpr size_t SZ_VW = (size_t)DN * DN;    // 1024x1024 transposed V-weights
constexpr size_t SZ_GW = (size_t)Dd * Dd;    // 256x256
constexpr size_t OFF_ENC = 0;
constexpr size_t BLK_BX = 0, BLK_AX = SZ_XW, BLK_TX = 2 * SZ_XW, BLK_IN = 3 * SZ_XW;
constexpr size_t BLK_BV = 4 * SZ_XW, BLK_AV = BLK_BV + SZ_VW, BLK_TV = BLK_BV + 2 * SZ_VW;
constexpr size_t BLK_GATE = BLK_BV + 3 * SZ_VW, BLK_SKIP = BLK_GATE + SZ_GW, BLK_OUT = BLK_SKIP + SZ_GW;
constexpr size_t BLK_SZ = BLK_OUT + SZ_XW;                       // per-block weight bundle
constexpr size_t OFF_BLK0 = SZ_GW;                               // after enc_Wt
constexpr size_t OFF_DEC = OFF_BLK0 + 2 * BLK_SZ;                // dec_Wt (VOCP x 256)
constexpr size_t HALF_TOTAL = OFF_DEC + (size_t)VOCP * Dd;
// ---- f32 state region (element offsets into float* after half region) ----
constexpr size_t F_H   = 0;                       // 8x256 encoder output
constexpr size_t F_FR  = F_H + 2048;              // 8 frames x 8x256 bits
constexpr size_t F_ZC  = F_FR + (size_t)Kk * 2048;// 4608 cols x 8 rows, column-major-by-8
constexpr size_t F_V   = F_ZC + (size_t)8 * 4608; // 2 x 8x1024 V state
constexpr size_t F_U   = F_V + (size_t)NBk * 8192;// 2 x 8x256 u state
constexpr size_t F_DEC = F_U + (size_t)NBk * 2048;// 64 x 8x256 decoded
}

#define DEV __device__ __forceinline__

DEV v8f wmma_f16(v16h a, v16h b, v8f c) {
  // D = A(16x32 f16) * B(32x16 f16) + C(16x16 f32)
  return __builtin_amdgcn_wmma_f32_16x16x32_f16(false, a, false, b, (short)0, c, false, false);
}

// A operand: 16 rows staged in LDS (row-major, stride lda halves), K window [k0, k0+32).
// A layout (contiguous k, matching load_b_gl): lanes 0-15 rows M=0..15 hold K = k0+{0..15},
// lanes 16-31 same rows hold K = k0+{16..31}.
DEV v16h load_a_lds(const _Float16* base, int lda, int k0) {
  int lane = threadIdx.x & 31;
  int row  = lane & 15;
  int kb   = k0 + ((lane >> 4) << 4);
  const _Float16* p = base + row * lda + kb;
  v16h a;
#pragma unroll
  for (int j = 0; j < 8; ++j) { a[j] = p[j]; a[8 + j] = p[8 + j]; }
  return a;
}

// B operand from global f16 weight stored transposed as (N, K) row-major:
// lane 0-15 -> col n0+lane, K = k0..k0+15 ; lane 16-31 -> same col, K = k0+16..k0+31.
// Each lane reads 32 contiguous bytes.
DEV v16h load_b_gl(const _Float16* Wt, int Kd, int n0, int k0) {
  int lane = threadIdx.x & 31;
  int col  = n0 + (lane & 15);
  int kb   = k0 + ((lane >> 4) << 4);
  const _Float16* p = Wt + (size_t)col * Kd + kb;
  v16h b;
#pragma unroll
  for (int j = 0; j < 16; ++j) b[j] = p[j];
  return b;
}

DEV float sigmoidf_(float x) { return 1.0f / (1.0f + __expf(-x)); }
DEV float softplusf_(float x) { return (x > 20.f) ? x : log1pf(__expf(x)); }

// -------- prep: convert f32 (K,N) row-major -> f16 (Npad,K) row-major (transposed) --------
__global__ void transpose_f16_k(const float* __restrict__ src, _Float16* __restrict__ dst,
                                int Kd, int N, int Npad) {
  long long total = (long long)Npad * Kd;
  for (long long idx = (long long)blockIdx.x * blockDim.x + threadIdx.x; idx < total;
       idx += (long long)gridDim.x * blockDim.x) {
    int n = (int)(idx / Kd), k = (int)(idx % Kd);
    float v = (n < N) ? src[(size_t)k * N + n] : 0.f;
    dst[idx] = (_Float16)v;
  }
}

// -------- persistent recurrent scan: one workgroup, 16 waves --------
__global__ __launch_bounds__(512)
void snn_recurrent(const int* __restrict__ tok_ids, const float* __restrict__ emb,
                   const float* __restrict__ enc_b,
                   const float* __restrict__ b_b, const float* __restrict__ b_a,
                   const float* __restrict__ b_t, const float* __restrict__ w_on,
                   const float* __restrict__ bitw,
                   const _Float16* __restrict__ hws, float* __restrict__ fws) {
  using namespace cfg;
  __shared__ _Float16 ldsA[16 * 1024];  // V-state / spike-s A operand (rows 8..15 zero pad)
  __shared__ _Float16 spA[16 * 256];    // sp A operand (rows 8..15 zero pad)

  const int tid = threadIdx.x, wave = tid >> 5, lane = tid & 31;
  const int NT = 512;

  float* H   = fws + F_H;
  float* FR  = fws + F_FR;
  float* ZC  = fws + F_ZC;
  float* V   = fws + F_V;
  float* U   = fws + F_U;
  float* DEC = fws + F_DEC;
  const _Float16* encWt = hws + OFF_ENC;

  // deterministic init of recurrent state
  for (int e = tid; e < NBk * 8192; e += NT) V[e] = 0.f;
  for (int e = tid; e < NBk * 2048; e += NT) U[e] = 0.f;
  __syncthreads();

  for (int s = 0; s < Ss; ++s) {
    // ---- encoder: stage emb rows for the 8 tokens (f16, rows 8..15 = 0) ----
    for (int idx = tid; idx < 16 * 256; idx += NT) {
      int row = idx >> 8, d = idx & 255;
      float v = 0.f;
      if (row < 8) { int t = tok_ids[row * Ss + s]; v = emb[(size_t)t * Dd + d]; }
      spA[idx] = (_Float16)v;
    }
    __syncthreads();
    {  // 16 tiles of 16 cols, one per wave; K=256 -> 8 WMMA
      int n0 = wave * 16;
      v8f acc = {};
      for (int k0 = 0; k0 < Dd; k0 += 32)
        acc = wmma_f16(load_a_lds(spA, Dd, k0), load_b_gl(encWt, Dd, n0, k0), acc);
      if (lane < 16) {
        int col = n0 + lane;
        float bb = enc_b[col];
#pragma unroll
        for (int r = 0; r < 8; ++r) H[r * Dd + col] = sigmoidf_(acc[r] + bb);
      }
    }
    __syncthreads();
    // ---- bit-plane frames ----
    for (int e = tid; e < 2048; e += NT) {
      float res = H[e];
#pragma unroll
      for (int k = 0; k < Kk; ++k) {
        float bit = (res >= 0.5f) ? 1.f : 0.f;
        FR[k * 2048 + e] = bit;
        res = (res - 0.5f * bit) * 2.f;
      }
    }
    __syncthreads();

    for (int k = 0; k < Kk; ++k) {
      // stage sp = frame k (rows 8..15 zero)
      for (int idx = tid; idx < 16 * 256; idx += NT) {
        int row = idx >> 8, d = idx & 255;
        spA[idx] = (_Float16)((row < 8) ? FR[k * 2048 + row * 256 + d] : 0.f);
      }
      __syncthreads();

      for (int blk = 0; blk < NBk; ++blk) {
        float* Vb = V + (size_t)blk * 8192;
        float* Ub = U + (size_t)blk * 2048;
        const _Float16* wb = hws + OFF_BLK0 + (size_t)blk * BLK_SZ;
        // stage V (f32 -> f16) as A operand, rows 8..15 zero
        for (int idx = tid; idx < 16 * 1024; idx += NT) {
          int row = idx >> 10, c = idx & 1023;
          ldsA[idx] = (_Float16)((row < 8) ? Vb[row * 1024 + c] : 0.f);
        }
        __syncthreads();

        // ---- Phase A: fused pre-activations over 4608 cols = [zb|za|zt|zin|zgate|zskip]
        const _Float16* Wx3[3] = { wb + BLK_BX, wb + BLK_AX, wb + BLK_TX };
        const _Float16* Wv3[3] = { wb + BLK_BV, wb + BLK_AV, wb + BLK_TV };
        for (int t = wave; t < 288; t += 16) {
          int n0 = t * 16;
          v8f acc = {};
          if (n0 < 3072) {               // beta/alpha/vth: sp*Wx + V*Wv
            int g = n0 >> 10, nc = n0 & 1023;
            const _Float16* Wx = Wx3[g];
            const _Float16* Wv = Wv3[g];
            for (int k0 = 0; k0 < 256;  k0 += 32)
              acc = wmma_f16(load_a_lds(spA, 256, k0),  load_b_gl(Wx, 256,  nc, k0), acc);
            for (int k0 = 0; k0 < 1024; k0 += 32)
              acc = wmma_f16(load_a_lds(ldsA, 1024, k0), load_b_gl(Wv, 1024, nc, k0), acc);
          } else if (n0 < 4096) {        // zin = sp*W_in
            int nc = n0 - 3072;
            for (int k0 = 0; k0 < 256; k0 += 32)
              acc = wmma_f16(load_a_lds(spA, 256, k0), load_b_gl(wb + BLK_IN, 256, nc, k0), acc);
          } else if (n0 < 4352) {        // zgate
            int nc = n0 - 4096;
            for (int k0 = 0; k0 < 256; k0 += 32)
              acc = wmma_f16(load_a_lds(spA, 256, k0), load_b_gl(wb + BLK_GATE, 256, nc, k0), acc);
          } else {                       // zskip
            int nc = n0 - 4352;
            for (int k0 = 0; k0 < 256; k0 += 32)
              acc = wmma_f16(load_a_lds(spA, 256, k0), load_b_gl(wb + BLK_SKIP, 256, nc, k0), acc);
          }
          if (lane < 16) {               // lanes 0-15 hold rows 0..7 (real batch)
            int col = n0 + lane;
#pragma unroll
            for (int r = 0; r < 8; ++r) ZC[(size_t)col * 8 + r] = acc[r];
          }
        }
        __syncthreads();

        // ---- Phase B: membrane update + spikes; write s into ldsA (A operand) ----
        for (int e = tid; e < 8192; e += NT) {
          int r = e & 7, c = e >> 3;
          float zb = ZC[e];
          float za = ZC[8192 + e];
          float zt = ZC[16384 + e];
          float zi = ZC[24576 + e];
          float beta  = sigmoidf_(zb + b_b[blk * 1024 + c]);
          float alpha = sigmoidf_(za + b_a[blk * 1024 + c]);
          float vth   = 0.1f + softplusf_(zt + b_t[blk * 1024 + c]);
          float vm = beta * Vb[r * 1024 + c] + alpha * zi;
          float sp = (vm >= vth) ? 1.f : 0.f;
          vm -= sp * vth;
          Vb[r * 1024 + c] = vm;
          ldsA[r * 1024 + c] = (_Float16)sp;  // rows 8..15 stay zero from staging
        }
        __syncthreads();

        // ---- Phase C: s @ W_out, gate, output LIF, new sp ----
        {
          int n0 = wave * 16;  // 256 cols -> 16 tiles, one per wave
          v8f acc = {};
          for (int k0 = 0; k0 < 1024; k0 += 32)
            acc = wmma_f16(load_a_lds(ldsA, 1024, k0), load_b_gl(wb + BLK_OUT, 1024, n0, k0), acc);
          if (lane < 16) {
            int col = n0 + lane;
            float won = sigmoidf_(w_on[blk * 256 + col]);
#pragma unroll
            for (int r = 0; r < 8; ++r) {
              float zg = ZC[(size_t)(4096 + col) * 8 + r];
              float zs = ZC[(size_t)(4352 + col) * 8 + r];
              float y  = sigmoidf_(zg) * acc[r] + zs;
              float uo = won * Ub[r * 256 + col] + y;
              float o  = (uo >= 1.f) ? 1.f : 0.f;
              Ub[r * 256 + col] = uo - o;
              spA[r * 256 + col] = (_Float16)o;   // feeds next block / decoded
            }
          }
        }
        __syncthreads();
      }  // blk

      // ---- decoded += bit_weights[k] * out_spikes ----
      {
        float bw = bitw[k];
        for (int e = tid; e < 2048; e += NT) {
          float o = (float)spA[e];
          float* dd = DEC + (size_t)s * 2048 + e;
          if (k == 0) *dd = bw * o; else *dd += bw * o;
        }
      }
      __syncthreads();
    }  // k frames
  }    // s tokens
}

// -------- decoder GEMM: (8x256 per token) x (256 x VOC), parallel over tokens/cols --------
__global__ __launch_bounds__(256)
void snn_decode(const float* __restrict__ dec_all, const _Float16* __restrict__ decWt,
                const float* __restrict__ dec_b, float* __restrict__ out) {
  using namespace cfg;
  __shared__ _Float16 aLds[16 * 256];
  int s = blockIdx.y;
  int tid = threadIdx.x, wave = tid >> 5, lane = tid & 31;
  for (int idx = tid; idx < 16 * 256; idx += 256) {
    int row = idx >> 8, d = idx & 255;
    aLds[idx] = (_Float16)((row < 8) ? dec_all[(size_t)s * 2048 + row * 256 + d] : 0.f);
  }
  __syncthreads();
  int t = blockIdx.x * 8 + wave;
  if (t < NDTILE) {
    int n0 = t * 16;
    v8f acc = {};
    for (int k0 = 0; k0 < 256; k0 += 32)
      acc = wmma_f16(load_a_lds(aLds, 256, k0), load_b_gl(decWt, 256, n0, k0), acc);
    if (lane < 16) {
      int col = n0 + lane;
      if (col < VOC) {
        float bb = dec_b[col];
#pragma unroll
        for (int r = 0; r < 8; ++r)
          out[((size_t)r * Ss + s) * VOC + col] = acc[r] + bb;  // (B,S,VOC)
      }
    }
  }
}

extern "C" void kernel_launch(void* const* d_in, const int* in_sizes, int n_in,
                              void* d_out, int out_size, void* d_ws, size_t ws_size,
                              hipStream_t stream) {
  using namespace cfg;
  const int*   token_ids = (const int*)  d_in[0];
  const float* emb    = (const float*)d_in[1];
  const float* enc_W  = (const float*)d_in[2];
  const float* enc_b  = (const float*)d_in[3];
  const float* dec_W  = (const float*)d_in[4];
  const float* dec_b  = (const float*)d_in[5];
  const float* bitw   = (const float*)d_in[6];
  const float* W_in   = (const float*)d_in[7];
  const float* W_bx   = (const float*)d_in[8];
  const float* W_ax   = (const float*)d_in[9];
  const float* W_tx   = (const float*)d_in[10];
  const float* W_bV   = (const float*)d_in[11];
  const float* W_aV   = (const float*)d_in[12];
  const float* W_tV   = (const float*)d_in[13];
  const float* b_b    = (const float*)d_in[14];
  const float* b_a    = (const float*)d_in[15];
  const float* b_t    = (const float*)d_in[16];
  const float* W_gate = (const float*)d_in[17];
  const float* W_skip = (const float*)d_in[18];
  const float* W_out  = (const float*)d_in[19];
  const float* w_on   = (const float*)d_in[20];

  _Float16* hws = (_Float16*)d_ws;
  float*    fws = (float*)((char*)d_ws + HALF_TOTAL * sizeof(_Float16));

  auto T = [&](const float* src, _Float16* dst, int Kd, int N, int Npad) {
    long long total = (long long)Npad * Kd;
    int blocks = (int)((total + 255) / 256);
    if (blocks > 4096) blocks = 4096;
    transpose_f16_k<<<blocks, 256, 0, stream>>>(src, dst, Kd, N, Npad);
  };

  // weight prep: transpose + f32->f16 (weights become L2-resident; total ~45 MB < 192 MB L2)
  T(enc_W, hws + OFF_ENC, Dd, Dd, Dd);
  for (int b2 = 0; b2 < NBk; ++b2) {
    _Float16* wb = hws + OFF_BLK0 + (size_t)b2 * BLK_SZ;
    T(W_bx   + (size_t)b2 * Dd * DN, wb + BLK_BX,   Dd, DN, DN);
    T(W_ax   + (size_t)b2 * Dd * DN, wb + BLK_AX,   Dd, DN, DN);
    T(W_tx   + (size_t)b2 * Dd * DN, wb + BLK_TX,   Dd, DN, DN);
    T(W_in   + (size_t)b2 * Dd * DN, wb + BLK_IN,   Dd, DN, DN);
    T(W_bV   + (size_t)b2 * DN * DN, wb + BLK_BV,   DN, DN, DN);
    T(W_aV   + (size_t)b2 * DN * DN, wb + BLK_AV,   DN, DN, DN);
    T(W_tV   + (size_t)b2 * DN * DN, wb + BLK_TV,   DN, DN, DN);
    T(W_gate + (size_t)b2 * Dd * Dd, wb + BLK_GATE, Dd, Dd, Dd);
    T(W_skip + (size_t)b2 * Dd * Dd, wb + BLK_SKIP, Dd, Dd, Dd);
    T(W_out  + (size_t)b2 * DN * Dd, wb + BLK_OUT,  DN, Dd, Dd);
  }
  T(dec_W, hws + OFF_DEC, Dd, VOC, VOCP);   // zero-padded to 50272 cols

  // persistent sequential scan (1 workgroup, 16 wave32 waves)
  snn_recurrent<<<1, 512, 0, stream>>>(token_ids, emb, enc_b, b_b, b_a, b_t, w_on, bitw,
                                       hws, fws);

  // parallel decoder GEMM over (tokens x vocab tiles)
  dim3 dg((NDTILE + 7) / 8, Ss);
  snn_decode<<<dg, 256, 0, stream>>>(fws + F_DEC, hws + OFF_DEC, dec_b, (float*)d_out);

  (void)in_sizes; (void)n_in; (void)out_size; (void)ws_size;
}
